// DepthConv_3788161155236
// MI455X (gfx1250) — matrix operand order repaired and checked by reading, not compiled, not run on hardware
//
#include <hip/hip_runtime.h>

typedef __attribute__((ext_vector_type(16))) _Float16 v16h;
typedef __attribute__((ext_vector_type(4)))  _Float16 v4h;
typedef __attribute__((ext_vector_type(8)))  float    v8f;

#define ALPHA 8.3f
#define CIN   32
#define COUT  32
#define HDIM  512
#define WDIM  512
#define TILE_W 128
#define XT_W  (TILE_W + 4)   // 132: 1-col halo each side + pad
#define CPAD  40             // padded channel stride (halves): 80B -> bank 20*l%64, conflict-free
#define WL    24             // padded A-frag lane stride (halves): 48B -> bank 12*l%64, conflict-free

#define WELEMS (9 * 2 * 32 * WL)     // padded packed-weight elements (halves) = 13824
#define WBYTES (WELEMS * 2)          // 27648 B

// Dynamic LDS layout (bytes):
//   wfrag : _Float16[9][2][32][WL]      pre-packed A fragments = 27648 B @ 0
//   xtile : _Float16[3][XT_W][CPAD]     f16 x halo tile        = 31680 B @ 27648
//   dtile : float[3][XT_W]              depth halo tile        =  1584 B @ 59328
#define WFRAG_OFF 0
#define XTILE_OFF 27648
#define DTILE_OFF 59328
#define LDS_TOTAL 60912

#define USE_ASYNC_WFRAG 1

// ---- One-shot: pack weights into per-lane WMMA A-fragment order (padded) ----
// A (16x32) as packed here, contiguous k as in B: lane half 0 -> K {0..15}; half 1 -> K {16..31}
__global__ void pack_weights_kernel(const float* __restrict__ weight,
                                    _Float16* __restrict__ wpack) {
    int idx = blockIdx.x * 256 + threadIdx.x;
    if (idx >= WELEMS) return;
    int e    = idx % WL;
    int lane = (idx / WL) % 32;
    int mt   = (idx / (WL * 32)) % 2;
    int t    = idx / (WL * 64);          // tap = ki*3 + kj
    _Float16 v = (_Float16)0.0f;
    if (e < 16) {
        int half = lane >> 4;
        int m    = lane & 15;
        int k    = half * 16 + e;
        int o    = mt * 16 + m;
        v = (_Float16)weight[(o * CIN + k) * 9 + t];
    }
    wpack[idx] = v;
}

__global__ void depthconv_wmma_f16(const float* __restrict__ x,
                                   const float* __restrict__ depth,
                                   const _Float16* __restrict__ wpack,
                                   const float* __restrict__ bias,
                                   float* __restrict__ out) {
    extern __shared__ char smem[];
    _Float16* wfrag = (_Float16*)(smem + WFRAG_OFF); // [9][2][32][WL]
    _Float16* xtile = (_Float16*)(smem + XTILE_OFF); // [3][XT_W][CPAD]
    float*    dtile = (float*)   (smem + DTILE_OFF); // [3][XT_W]

    const int tid = threadIdx.x;          // 256 threads = 8 waves (wave32)
    const int w0  = blockIdx.x * TILE_W;
    const int h   = blockIdx.y;
    const int b   = blockIdx.z;

    // ---- Stage pre-packed A fragments: contiguous global->LDS copy ----
#if USE_ASYNC_WFRAG
    {
        // Per-lane async DMA, 16B per lane per op; tracked by ASYNCcnt.
        unsigned lds0 = (unsigned)(size_t)(void*)wfrag;   // LDS byte address (low 32 bits)
        for (int idx = tid; idx < WBYTES / 16; idx += 256) {
            unsigned goff = (unsigned)idx * 16u;          // byte offset from SGPR base
            unsigned ldsa = lds0 + goff;
            asm volatile("global_load_async_to_lds_b128 %0, %1, %2"
                         :: "v"(ldsa), "v"(goff), "s"(wpack) : "memory");
        }
        asm volatile("s_wait_asynccnt 0" ::: "memory");
    }
#else
    for (int idx = tid; idx < WBYTES / 16; idx += 256)
        ((uint4*)wfrag)[idx] = ((const uint4*)wpack)[idx];
#endif

    // ---- Stage x halo tile as f16, channel-fastest; 4 channels/thread ----
    // idx: j fastest for coalesced global reads; one ds_store_b64 per unit.
    for (int idx = tid; idx < (CIN / 4) * 3 * XT_W; idx += 256) {
        int j  = idx % XT_W;
        int r  = (idx / XT_W) % 3;
        int cq = idx / (3 * XT_W);        // channel quad
        int gh = h - 1 + r;
        int gw = w0 - 1 + j;
        float v0 = 0.0f, v1 = 0.0f, v2 = 0.0f, v3 = 0.0f;
        if ((unsigned)gh < HDIM && (unsigned)gw < WDIM) {
            const float* xp = &x[((b * CIN + cq * 4) * HDIM + gh) * WDIM + gw];
            v0 = xp[0 * HDIM * WDIM];
            v1 = xp[1 * HDIM * WDIM];
            v2 = xp[2 * HDIM * WDIM];
            v3 = xp[3 * HDIM * WDIM];
        }
        v4h pk = { (_Float16)v0, (_Float16)v1, (_Float16)v2, (_Float16)v3 };
        *(v4h*)&xtile[(r * XT_W + j) * CPAD + cq * 4] = pk;   // 8B-aligned b64 store
    }

    // ---- Stage depth halo tile (zero-padded, fp32) ----
    for (int idx = tid; idx < 3 * XT_W; idx += 256) {
        int j  = idx % XT_W;
        int r  = idx / XT_W;
        int gh = h - 1 + r;
        int gw = w0 - 1 + j;
        float v = 0.0f;
        if ((unsigned)gh < HDIM && (unsigned)gw < WDIM)
            v = depth[(b * HDIM + gh) * WDIM + gw];
        dtile[idx] = v;
    }

    __syncthreads();

    // ---- Per-wave GEMM: 32 (C_out) x 16 (pixels) tile, K = 9 taps * 32 ch ----
    const int wave = tid >> 5;
    const int lane = tid & 31;
    const int half = lane >> 4;
    const int col  = lane & 15;
    const int px   = wave * 16 + col;     // pixel within 128-wide tile

    // C/D layout: lanes 0-15 VGPR r -> M=r ; lanes 16-31 -> M=8+r. Init with bias (b128 loads).
    const float4 b0 = *(const float4*)&bias[half * 8];
    const float4 b1 = *(const float4*)&bias[half * 8 + 4];
    const float4 b2 = *(const float4*)&bias[16 + half * 8];
    const float4 b3 = *(const float4*)&bias[16 + half * 8 + 4];
    v8f acc0 = { b0.x, b0.y, b0.z, b0.w, b1.x, b1.y, b1.z, b1.w };
    v8f acc1 = { b2.x, b2.y, b2.z, b2.w, b3.x, b3.y, b3.z, b3.w };

    const float dc    = dtile[1 * XT_W + 1 + px];   // center depth for this pixel
    const int   kbase = half * 16;                  // B k order used here (same as A pack): half 0 -> K 0..15, half 1 -> K 16..31

#pragma unroll
    for (int t = 0; t < 9; ++t) {
        const int ki = t / 3, kj = t % 3;

        float dsv = dtile[ki * XT_W + kj + px];
        float sim = __expf(-ALPHA * fabsf(dc - dsv));   // trans pipe, co-executes with WMMA
        _Float16 simh = (_Float16)sim;

        // B fragment: 16 contiguous halves (channels kbase..kbase+15) at this tap's pixel,
        // broadcast-modulated in packed f16.
        const v16h xv = *(const v16h*)&xtile[(ki * XT_W + kj + px) * CPAD + kbase];
        v16h bfrag = xv * simh;                          // 8x v_pk_mul_f16

        v16h a0 = *(const v16h*)&wfrag[((t * 2 + 0) * 32 + lane) * WL];
        v16h a1 = *(const v16h*)&wfrag[((t * 2 + 1) * 32 + lane) * WL];

        acc0 = __builtin_amdgcn_wmma_f32_16x16x32_f16(false, a0, false, bfrag,
                                                      (short)0, acc0, false, false);
        acc1 = __builtin_amdgcn_wmma_f32_16x16x32_f16(false, a1, false, bfrag,
                                                      (short)0, acc1, false, false);
    }

    // ---- Store: D layout -> out[b][o][h][w0+px] ----
    const int gw = w0 + px;
#pragma unroll
    for (int r = 0; r < 8; ++r) {
        int o0 = half * 8 + r;
        int o1 = 16 + half * 8 + r;
        out[((b * COUT + o0) * HDIM + h) * WDIM + gw] = acc0[r];
        out[((b * COUT + o1) * HDIM + h) * WDIM + gw] = acc1[r];
    }
}

extern "C" void kernel_launch(void* const* d_in, const int* in_sizes, int n_in,
                              void* d_out, int out_size, void* d_ws, size_t ws_size,
                              hipStream_t stream) {
    (void)in_sizes; (void)n_in; (void)out_size; (void)ws_size;
    const float* x      = (const float*)d_in[0];
    const float* depth  = (const float*)d_in[1];
    const float* weight = (const float*)d_in[2];
    const float* bias   = (const float*)d_in[3];
    float* out = (float*)d_out;
    _Float16* wpack = (_Float16*)d_ws;    // 27648 B of scratch

    // One-shot fragment packing (cheap; runs every launch for determinism).
    hipLaunchKernelGGL(pack_weights_kernel, dim3((WELEMS + 255) / 256), dim3(256),
                       0, stream, weight, wpack);

    dim3 grid(WDIM / TILE_W, HDIM, 2);   // 4 x 512 x 2 = 4096 workgroups
    dim3 block(256);                     // 8 waves (wave32)
    hipLaunchKernelGGL(depthconv_wmma_f16, grid, block, LDS_TOTAL, stream,
                       x, depth, wpack, bias, out);
}
